// DeepseekV2LiteAttention_13675175870990
// MI455X (gfx1250) — matrix operand order repaired and checked by reading, not compiled, not run on hardware
//
#include <hip/hip_runtime.h>
#include <hip/hip_bf16.h>
#include <math.h>

// ---------------------------------------------------------------- types
typedef __bf16 bf16;
typedef __attribute__((ext_vector_type(16))) __bf16 v16bf;
typedef __attribute__((ext_vector_type(8)))  __bf16 v8bf;
typedef __attribute__((ext_vector_type(8)))  float  v8f;

#define B_      2
#define S_      2048
#define HID_    2048
#define NH_     16
#define D_NOPE_ 128
#define D_ROPE_ 64
#define D_V_    128
#define D_Q_    192          // D_NOPE + D_ROPE
#define KV_RANK_ 512
#define TOK_    (B_*S_)      // 4096
#define QB_     128          // queries per flash-attention block (8 waves x 16)

#if __has_builtin(__builtin_amdgcn_tensor_load_to_lds) && __has_builtin(__builtin_amdgcn_s_wait_tensorcnt)
#define USE_TDM 1
#else
#define USE_TDM 0
#endif

__device__ __forceinline__ v8f wmma_bf16(v16bf a, v16bf b, v8f c) {
  return __builtin_amdgcn_wmma_f32_16x16x32_bf16(false, a, false, b, (short)0, c, false, false);
}

// ---------------------------------------------------------------- fp32 -> bf16
__global__ void k_f32_to_bf16(const float* __restrict__ src, bf16* __restrict__ dst, int n) {
  int i = blockIdx.x * blockDim.x + threadIdx.x;
  int stride = gridDim.x * blockDim.x;
  for (; i < n; i += stride) dst[i] = (bf16)src[i];
}

// ---------------------------------------------------------------- generic WMMA GEMM
// C[M,N] = A[M,K](bf16, row-major) * W[N,K](bf16, row-major)^T
// block = 128 threads (4 waves); wave computes a 32x64 tile (2 A-frags share B-frags);
// grid = (N/64, M/128)
__launch_bounds__(128)
__global__ void k_gemm_bf16(const bf16* __restrict__ A, const bf16* __restrict__ W,
                            void* __restrict__ C, int M, int N, int K, int store_bf16) {
  const int lane = threadIdx.x & 31;
  const int wave = threadIdx.x >> 5;
  const int half = lane >> 4;
  const int r    = lane & 15;
  const int m0 = blockIdx.y * 128 + wave * 32;
  const int n0 = blockIdx.x * 64;

  v8f acc[2][4] = {};
  const bf16* arow0 = A + (size_t)(m0 + r) * K;
  const bf16* arow1 = A + (size_t)(m0 + 16 + r) * K;

  for (int kb = 0; kb < K; kb += 32) {
    __builtin_prefetch(arow0 + kb + 256, 0, 1);   // global_prefetch_b8, stream-ahead
    // A fragment: lane holds 16 contiguous K at kb + half*16
    v8bf a0lo = *(const v8bf*)(arow0 + kb + half * 16);
    v8bf a0hi = *(const v8bf*)(arow0 + kb + 8 + half * 16);
    v8bf a1lo = *(const v8bf*)(arow1 + kb + half * 16);
    v8bf a1hi = *(const v8bf*)(arow1 + kb + 8 + half * 16);
    v16bf af0, af1;
    #pragma unroll
    for (int i = 0; i < 8; ++i) {
      af0[i] = a0lo[i]; af0[8 + i] = a0hi[i];
      af1[i] = a1lo[i]; af1[8 + i] = a1hi[i];
    }
    #pragma unroll
    for (int t = 0; t < 4; ++t) {
      // B fragment: lane's column n = n0+t*16+r, 16 contiguous K at kb + half*16
      v16bf bfrag = *(const v16bf*)(W + (size_t)(n0 + t * 16 + r) * K + kb + half * 16);
      acc[0][t] = wmma_bf16(af0, bfrag, acc[0][t]);
      acc[1][t] = wmma_bf16(af1, bfrag, acc[1][t]);
    }
  }

  #pragma unroll
  for (int u = 0; u < 2; ++u)
    #pragma unroll
    for (int t = 0; t < 4; ++t)
      #pragma unroll
      for (int j = 0; j < 8; ++j) {
        int row = m0 + u * 16 + j + 8 * half;   // C layout: VGPR j -> M = j (+8 upper lanes)
        int col = n0 + t * 16 + r;              // N = lane & 15
        float v = acc[u][t][j];
        if (store_bf16) ((bf16*)C)[(size_t)row * N + col] = (bf16)v;
        else            ((float*)C)[(size_t)row * N + col] = v;
      }
}

// ---------------------------------------------------------------- YaRN inv_freq
__device__ __forceinline__ float yarn_inv_freq(int j) {
  const float dim = 64.f, base = 10000.f, factor = 40.f, orig = 4096.f;
  const float two_pi = 6.283185307f;
  float fe = __powf(base, -(2.f * j) / dim);   // freq_extra
  float fi = fe / factor;                      // freq_inter
  float low  = floorf(dim * __logf(orig / (32.f * two_pi)) / (2.f * __logf(base)));
  float high = ceilf (dim * __logf(orig / ( 1.f * two_pi)) / (2.f * __logf(base)));
  low = fmaxf(low, 0.f); high = fminf(high, dim - 1.f);
  float hi = (low == high) ? high + 0.001f : high;
  float ramp = fminf(fmaxf(((float)j - low) / (hi - low), 0.f), 1.f);
  float mask = 1.f - ramp;
  return fi * (1.f - mask) + fe * mask;        // mscale ratio == 1.0 (MSCALE==MSCALE_ALL)
}

// ---------------------------------------------------------------- RMSNorm + k_pe RoPE
__global__ void k_rmsnorm_kpe(const float* __restrict__ ckv, const float* __restrict__ lnw,
                              const int* __restrict__ pos, bf16* __restrict__ cln,
                              bf16* __restrict__ kpe) {
  const int tok = blockIdx.x;
  const float* row = ckv + (size_t)tok * (KV_RANK_ + D_ROPE_);
  __shared__ float red[8];
  __shared__ float rs_sh;

  float ss = 0.f;
  for (int i = threadIdx.x; i < KV_RANK_; i += 256) { float v = row[i]; ss += v * v; }
  #pragma unroll
  for (int off = 16; off > 0; off >>= 1) ss += __shfl_xor(ss, off, 32);
  if ((threadIdx.x & 31) == 0) red[threadIdx.x >> 5] = ss;
  __syncthreads();
  if (threadIdx.x == 0) {
    float t = 0.f;
    #pragma unroll
    for (int i = 0; i < 8; ++i) t += red[i];
    rs_sh = rsqrtf(t / (float)KV_RANK_ + 1e-6f);
  }
  __syncthreads();
  const float rs = rs_sh;
  for (int i = threadIdx.x; i < KV_RANK_; i += 256)
    cln[(size_t)tok * KV_RANK_ + i] = (bf16)(row[i] * rs * lnw[i]);

  if (threadIdx.x < 32) {
    int j = threadIdx.x;
    float p = (float)pos[tok];
    float sn, cs; __sincosf(p * yarn_inv_freq(j), &sn, &cs);
    float x0 = row[KV_RANK_ + 2 * j];
    float x1 = row[KV_RANK_ + 2 * j + 1];
    kpe[(size_t)tok * 64 + j]      = (bf16)(x0 * cs - x1 * sn);
    kpe[(size_t)tok * 64 + 32 + j] = (bf16)(x1 * cs + x0 * sn);
  }
}

// ---------------------------------------------------------------- build qf [b,h,s,192] bf16
__global__ void k_build_qf(const float* __restrict__ q, const int* __restrict__ pos,
                           bf16* __restrict__ qf) {
  const size_t total = (size_t)TOK_ * NH_ * 96;
  for (size_t idx = (size_t)blockIdx.x * blockDim.x + threadIdx.x; idx < total;
       idx += (size_t)gridDim.x * blockDim.x) {
    int j = (int)(idx % 96);
    size_t th = idx / 96;
    int h = (int)(th % NH_);
    int tok = (int)(th / NH_);
    int b = tok / S_, s = tok % S_;
    bf16* dst = qf + ((size_t)(b * NH_ + h) * S_ + s) * D_Q_;
    const float* src = q + (size_t)tok * (NH_ * D_Q_) + h * D_Q_;
    if (j < 64) {
      dst[j]      = (bf16)src[j];
      dst[j + 64] = (bf16)src[j + 64];
    } else {
      int jj = j - 64;
      float p = (float)pos[tok];
      float sn, cs; __sincosf(p * yarn_inv_freq(jj), &sn, &cs);
      float x0 = src[128 + 2 * jj], x1 = src[128 + 2 * jj + 1];
      dst[128 + jj] = (bf16)(x0 * cs - x1 * sn);
      dst[160 + jj] = (bf16)(x1 * cs + x0 * sn);
    }
  }
}

// ---------------------------------------------------------------- build kf [b,h,s,192] and vT [b,h,d,s]
__global__ void k_build_kf_vt(const bf16* __restrict__ kv, const bf16* __restrict__ kpe,
                              bf16* __restrict__ kf, bf16* __restrict__ vT) {
  const size_t total = (size_t)TOK_ * NH_ * 320;
  for (size_t idx = (size_t)blockIdx.x * blockDim.x + threadIdx.x; idx < total;
       idx += (size_t)gridDim.x * blockDim.x) {
    int d = (int)(idx % 320);
    size_t th = idx / 320;
    int h = (int)(th % NH_);
    int tok = (int)(th / NH_);
    int b = tok / S_, s = tok % S_;
    size_t bh = (size_t)b * NH_ + h;
    if (d < 192) {
      bf16 v = (d < 128) ? kv[(size_t)tok * 4096 + h * 256 + d]
                         : kpe[(size_t)tok * 64 + (d - 128)];
      kf[(bh * S_ + s) * D_Q_ + d] = v;
    } else {
      int dv = d - 192;
      vT[(bh * D_V_ + dv) * S_ + s] = kv[(size_t)tok * 4096 + h * 256 + 128 + dv];
    }
  }
}

// ---------------------------------------------------------------- TDM 2-D tile load
#if USE_TDM
typedef __attribute__((ext_vector_type(4))) unsigned int u32x4;
typedef __attribute__((ext_vector_type(8))) int          i32x8;
typedef __attribute__((ext_vector_type(4))) int          i32x4;

// D# per CDNA5 ISA: 2-D tensor, data_size=2B, no pad/iterate.
__device__ __forceinline__ void tdm_load_2d(unsigned lds_byte_addr, const bf16* gsrc,
                                            unsigned td0, unsigned td1,
                                            unsigned tile0, unsigned tile1,
                                            unsigned stride0_elems) {
  unsigned long long ga = (unsigned long long)(const void*)gsrc;
  u32x4 g0;
  g0[0] = 1u;                                                  // count=1 (valid user D#)
  g0[1] = lds_byte_addr;                                       // lds_addr
  g0[2] = (unsigned)ga;                                        // global_addr[31:0]
  g0[3] = (unsigned)((ga >> 32) & 0x01FFFFFFu) | (2u << 30);   // global_addr[56:32] | type=2
  i32x8 g1;
  g1[0] = (int)(1u << 16);                                     // wg_mask=0, data_size=2B
  g1[1] = (int)(td0 << 16);                                    // tensor_dim0[15:0]
  g1[2] = (int)((td0 >> 16) | (td1 << 16));                    // tensor_dim0[31:16] | tensor_dim1[15:0]
  g1[3] = (int)((td1 >> 16) | (tile0 << 16));                  // tensor_dim1[31:16] | tile_dim0
  g1[4] = (int)(tile1 & 0xFFFFu);                              // tile_dim1 | tile_dim2=0
  g1[5] = (int)stride0_elems;                                  // tensor_dim0_stride[31:0]
  g1[6] = 0;                                                   // stride0 hi | stride1 lo
  g1[7] = 0;                                                   // stride1 hi
  i32x4 z4 = {0, 0, 0, 0};
#if __clang_major__ >= 23
  i32x8 z8 = {0, 0, 0, 0, 0, 0, 0, 0};
  __builtin_amdgcn_tensor_load_to_lds(g0, g1, z4, z4, z8, 0);
#else
  __builtin_amdgcn_tensor_load_to_lds(g0, g1, z4, z4, 0);
#endif
}
#endif

// ---------------------------------------------------------------- flash attention
// grid = (B*NH, S/QB_), block = 256 (8 waves). Block covers 128 contiguous queries of
// one (b,h); K/V tiles (32 keys) are TDM-staged into LDS, double-buffered, shared by all
// 8 waves. Uniform trip count per block -> barriers are safe.
__launch_bounds__(256)
__global__ void k_flash_attn(const bf16* __restrict__ qf, const bf16* __restrict__ kf,
                             const bf16* __restrict__ vT, bf16* __restrict__ aout) {
  __shared__ __align__(32) bf16 kbuf[2][32 * D_Q_];   // 24 KiB
  __shared__ __align__(32) bf16 vbuf[2][D_V_ * 32];   // 16 KiB
  __shared__ __align__(32) bf16 pbuf[8][16 * 32];     //  8 KiB (per-wave P restage)
  const int tid  = threadIdx.x;
  const int lane = tid & 31;
  const int wave = tid >> 5;
  const int half = lane >> 4;
  const int r    = lane & 15;
  const int bh   = blockIdx.x;
  const int qblock = blockIdx.y * QB_;
  const int qbase  = qblock + wave * 16;
  const float scale = 0.07216878365f;                 // 1/sqrt(192)

  // Q fragments (16 rows x 192, six K=32 slabs)
  const bf16* qrow = qf + ((size_t)bh * S_ + qbase + r) * D_Q_;
  v16bf aq[6];
  #pragma unroll
  for (int kb = 0; kb < 6; ++kb) {
    v8bf lo = *(const v8bf*)(qrow + kb * 32 + half * 16);
    v8bf hi = *(const v8bf*)(qrow + kb * 32 + 8 + half * 16);
    #pragma unroll
    for (int i = 0; i < 8; ++i) { aq[kb][i] = lo[i]; aq[kb][8 + i] = hi[i]; }
  }

  v8f o[8] = {};
  float mrow[8], lrow[8];
  #pragma unroll
  for (int j = 0; j < 8; ++j) { mrow[j] = -1e30f; lrow[j] = 0.f; }
  bf16* pp = &pbuf[wave][0];

  const int niter = (qblock + QB_) / 32;

  auto stage = [&](int it) {
    const int k0 = it * 32, buf = it & 1;
#if USE_TDM
    if (wave == 0) {
      tdm_load_2d((unsigned)(unsigned long long)(void*)&kbuf[buf][0],
                  kf + ((size_t)bh * S_ + k0) * D_Q_,
                  D_Q_, S_, D_Q_, 32, D_Q_);                 // 32 key rows x 192
      tdm_load_2d((unsigned)(unsigned long long)(void*)&vbuf[buf][0],
                  vT + (size_t)bh * D_V_ * S_ + k0,
                  S_, D_V_, 32, D_V_, S_);                   // 128 d rows x 32 keys
    }
#else
    const v8bf* ks = (const v8bf*)(kf + ((size_t)bh * S_ + k0) * D_Q_);
    v8bf* kd = (v8bf*)&kbuf[buf][0];
    for (int c = tid; c < 32 * D_Q_ / 8; c += 256) kd[c] = ks[c];
    v8bf* vd = (v8bf*)&vbuf[buf][0];
    for (int c = tid; c < D_V_ * 32 / 8; c += 256) {
      int row = c >> 2, off = (c & 3) * 8;
      vd[c] = *(const v8bf*)(vT + ((size_t)bh * D_V_ + row) * S_ + k0 + off);
    }
#endif
  };

  stage(0);
  for (int it = 0; it < niter; ++it) {
    const int k0 = it * 32, buf = it & 1;
    const bool more = (it + 1 < niter);
    if (more) stage(it + 1);                     // overlap DMA of next tile with compute
#if USE_TDM
    if (wave == 0) {
      if (more) __builtin_amdgcn_s_wait_tensorcnt(2);   // tile(it) done, tile(it+1) in flight
      else      __builtin_amdgcn_s_wait_tensorcnt(0);
    }
#endif
    __syncthreads();                             // tile(it) visible to all waves
    const bf16* kb_ = &kbuf[buf][0];
    const bf16* vb_ = &vbuf[buf][0];

    // ---- S = Q K^T, two 16x16 key subtiles (from LDS)
    v8f s0 = {}, s1 = {};
    #pragma unroll
    for (int kb = 0; kb < 6; ++kb) {
      v16bf b0 = *(const v16bf*)(kb_ + r * D_Q_        + kb * 32 + half * 16);
      v16bf b1 = *(const v16bf*)(kb_ + (16 + r) * D_Q_ + kb * 32 + half * 16);
      s0 = wmma_bf16(aq[kb], b0, s0);
      s1 = wmma_bf16(aq[kb], b1, s1);
    }
    const bool need_mask = (k0 + 31 > qbase);    // true for diag & fully-future tiles
    // ---- online softmax (row across 16 lanes; rows j+8*half per lane)
    #pragma unroll
    for (int j = 0; j < 8; ++j) {
      int rowg = qbase + j + 8 * half;
      float v0 = s0[j] * scale, v1 = s1[j] * scale;
      if (need_mask) {
        if (k0 + r      > rowg) v0 = -1e30f;
        if (k0 + 16 + r > rowg) v1 = -1e30f;
      }
      float mx = fmaxf(v0, v1);
      #pragma unroll
      for (int off = 8; off; off >>= 1) mx = fmaxf(mx, __shfl_xor(mx, off, 32));
      float mnew  = fmaxf(mrow[j], mx);
      float alpha = __expf(mrow[j] - mnew);
      float p0 = __expf(v0 - mnew);
      float p1 = __expf(v1 - mnew);
      float ps = p0 + p1;
      #pragma unroll
      for (int off = 8; off; off >>= 1) ps += __shfl_xor(ps, off, 32);
      lrow[j] = lrow[j] * alpha + ps;
      mrow[j] = mnew;
      #pragma unroll
      for (int t = 0; t < 8; ++t) o[t][j] *= alpha;
      pp[(j + 8 * half) * 32 + r]      = (bf16)p0;
      pp[(j + 8 * half) * 32 + 16 + r] = (bf16)p1;
    }
    // ---- reload P in A-fragment layout (wave-local LDS, DScnt keeps order)
    v8bf plo = *(const v8bf*)(pp + r * 32 + half * 16);
    v8bf phi = *(const v8bf*)(pp + r * 32 + 8 + half * 16);
    v16bf pa;
    #pragma unroll
    for (int i = 0; i < 8; ++i) { pa[i] = plo[i]; pa[8 + i] = phi[i]; }
    // ---- O += P V  (V tile from LDS, contiguous along keys)
    #pragma unroll
    for (int t = 0; t < 8; ++t) {
      v16bf bv = *(const v16bf*)(vb_ + (t * 16 + r) * 32 + half * 16);
      o[t] = wmma_bf16(pa, bv, o[t]);
    }
    __syncthreads();                             // all waves done reading tile(it)
  }

  // ---- epilogue: out[b, q, h*128+d] bf16
  const int b = bh / NH_, h = bh % NH_;
  #pragma unroll
  for (int t = 0; t < 8; ++t)
    #pragma unroll
    for (int j = 0; j < 8; ++j) {
      int rowg = qbase + j + 8 * half;
      int col  = h * D_V_ + t * 16 + r;
      aout[((size_t)b * S_ + rowg) * (NH_ * D_V_) + col] = (bf16)(o[t][j] / lrow[j]);
    }
}

// ---------------------------------------------------------------- host launch
extern "C" void kernel_launch(void* const* d_in, const int* in_sizes, int n_in,
                              void* d_out, int out_size, void* d_ws, size_t ws_size,
                              hipStream_t stream) {
  const float* hs      = (const float*)d_in[0];
  const int*   pos     = (const int*)  d_in[1];
  const float* q_w     = (const float*)d_in[2];
  const float* kv_a_w  = (const float*)d_in[3];
  const float* kv_ln_w = (const float*)d_in[4];
  const float* kv_b_w  = (const float*)d_in[5];
  const float* o_w     = (const float*)d_in[6];
  float* out = (float*)d_out;

  // ---- workspace carve-up
  char* p = (char*)d_ws;
  auto take = [&](size_t bytes) { char* q0 = p; p += (bytes + 255) & ~(size_t)255; return (void*)q0; };
  bf16* hs_bf   = (bf16*)take((size_t)TOK_ * HID_ * 2);
  bf16* qw_bf   = (bf16*)take((size_t)NH_ * D_Q_ * HID_ * 2);
  bf16* kvaw_bf = (bf16*)take((size_t)(KV_RANK_ + D_ROPE_) * HID_ * 2);
  bf16* kvbw_bf = (bf16*)take((size_t)NH_ * 256 * KV_RANK_ * 2);
  bf16* ow_bf   = (bf16*)take((size_t)HID_ * NH_ * D_V_ * 2);
  float* q_f32  = (float*)take((size_t)TOK_ * NH_ * D_Q_ * 4);   // reused as attn-out
  float* ckv_f  = (float*)take((size_t)TOK_ * (KV_RANK_ + D_ROPE_) * 4);
  bf16* cln_bf  = (bf16*)take((size_t)TOK_ * KV_RANK_ * 2);
  bf16* kpe_bf  = (bf16*)take((size_t)TOK_ * 64 * 2);
  bf16* kv_bf   = (bf16*)take((size_t)TOK_ * NH_ * 256 * 2);
  bf16* qf_bf   = (bf16*)take((size_t)B_ * NH_ * S_ * D_Q_ * 2);
  bf16* kf_bf   = (bf16*)take((size_t)B_ * NH_ * S_ * D_Q_ * 2);
  bf16* vT_bf   = (bf16*)take((size_t)B_ * NH_ * D_V_ * S_ * 2);
  bf16* aout_bf = (bf16*)q_f32;                                  // alias: q_f32 dead by then

  // ---- 1. fp32 -> bf16 conversions
  k_f32_to_bf16<<<2048, 256, 0, stream>>>(hs,     hs_bf,   TOK_ * HID_);
  k_f32_to_bf16<<<2048, 256, 0, stream>>>(q_w,    qw_bf,   NH_ * D_Q_ * HID_);
  k_f32_to_bf16<<<1024, 256, 0, stream>>>(kv_a_w, kvaw_bf, (KV_RANK_ + D_ROPE_) * HID_);
  k_f32_to_bf16<<<1024, 256, 0, stream>>>(kv_b_w, kvbw_bf, NH_ * 256 * KV_RANK_);
  k_f32_to_bf16<<<2048, 256, 0, stream>>>(o_w,    ow_bf,   HID_ * NH_ * D_V_);

  // ---- 2. q = hs * q_w^T  (4096x3072, K=2048), f32
  k_gemm_bf16<<<dim3(3072 / 64, TOK_ / 128), 128, 0, stream>>>(
      hs_bf, qw_bf, q_f32, TOK_, 3072, HID_, 0);
  // ---- 3. ckv = hs * kv_a_w^T  (4096x576), f32
  k_gemm_bf16<<<dim3(576 / 64, TOK_ / 128), 128, 0, stream>>>(
      hs_bf, kvaw_bf, ckv_f, TOK_, 576, HID_, 0);

  // ---- 4. RMSNorm + k_pe RoPE
  k_rmsnorm_kpe<<<TOK_, 256, 0, stream>>>(ckv_f, kv_ln_w, pos, cln_bf, kpe_bf);

  // ---- 5. kv = c_ln * kv_b_w^T  (4096x4096, K=512), bf16
  k_gemm_bf16<<<dim3(4096 / 64, TOK_ / 128), 128, 0, stream>>>(
      cln_bf, kvbw_bf, kv_bf, TOK_, NH_ * 256, KV_RANK_, 1);

  // ---- 6. build qf (RoPE on q_pe) and kf / vT
  k_build_qf<<<4096, 256, 0, stream>>>(q_f32, pos, qf_bf);
  k_build_kf_vt<<<8192, 256, 0, stream>>>(kv_bf, kpe_bf, kf_bf, vT_bf);

  // ---- 7. flash attention (TDM-staged K/V tiles in LDS, double-buffered)
  k_flash_attn<<<dim3(B_ * NH_, S_ / QB_), 256, 0, stream>>>(qf_bf, kf_bf, vT_bf, aout_bf);

  // ---- 8. final o-proj: out = aout * o_w^T (4096x2048, K=2048), f32 -> d_out
  k_gemm_bf16<<<dim3(2048 / 64, TOK_ / 128), 128, 0, stream>>>(
      aout_bf, ow_bf, out, TOK_, HID_, NH_ * D_V_, 0);

  (void)in_sizes; (void)n_in; (void)out_size; (void)ws_size;
}
